// MambaImageClassifier_55448027791791
// MI455X (gfx1250) — matrix operand order repaired and checked by reading, not compiled, not run on hardware
//
#include <hip/hip_runtime.h>
#include <hip/hip_bf16.h>
#include <stdint.h>

// ---------------------------------------------------------------------------
// Mamba image classifier for gfx1250 (MI455X).
// Large GEMMs: double-buffered async global->LDS staging
// (GLOBAL_LOAD_ASYNC_TO_LDS_B128 + s_wait_asynccnt 4 overlap) feeding
// v_wmma_f32_16x16x32_bf16 from ds_load_b128 fragments.
// Edge GEMM (head) uses a guarded global-direct WMMA kernel.
// ---------------------------------------------------------------------------

#define BB 16
#define LL 196
#define MTOK (BB * LL)      // 3136 token rows
#define DM 512
#define DI 1024
#define DTR 32
#define DSTATE 16
#define NCONV 4
#define NLAYERS 4
#define NCLASSES 1000
#define PATCHD 768
#define NPS 14
#define PHH 16

typedef __attribute__((ext_vector_type(16))) __bf16 v16bf;
typedef __attribute__((ext_vector_type(8)))  float  v8f;
typedef __attribute__((ext_vector_type(4)))  unsigned int u32x4;

union FragU {
  v16bf v;
  u32x4 q[2];
  unsigned int u[8];
};

__device__ __forceinline__ unsigned short f2bf(float f) {
  unsigned int u = __float_as_uint(f);
  unsigned int r = u + 0x7fffu + ((u >> 16) & 1u);   // round-to-nearest-even
  return (unsigned short)(r >> 16);
}

// ---------------------------------------------------------------------------
// LDS-staged WMMA GEMM:  C[M,N] = A[M,K] * B (Bt[N,K] bf16, B transposed)
// Requires M%64==0, N%64==0, K%32==0 (all big GEMMs here qualify).
// Block = 128 threads (4 waves); block tile 64x64; wave tile 32x32 (2x2 WMMA).
// Double-buffered: tile k+1 is async-copied while tile k is multiplied.
// ---------------------------------------------------------------------------
#define TM 64
#define TN 64
#define TK 32

// Per-thread share of one 64x32 bf16 tile pair: 4 async b128 copies
// (2 chunks of A + 2 chunks of Bt), 256 chunks of 16B per tile.
__device__ __forceinline__ void stage_tiles(const unsigned short* __restrict__ A,
                                            const unsigned short* __restrict__ Bt,
                                            int K, int m0, int n0, int k0,
                                            int tid, unsigned int ldsA,
                                            unsigned int ldsB) {
#pragma unroll
  for (int cc = 0; cc < 2; ++cc) {
    int c   = tid + cc * 128;            // 0..255
    int row = c >> 2;                    // 0..63
    int col = (c & 3) * 8;               // element offset inside 32-wide K block
    const unsigned short* ga = A  + (size_t)(m0 + row) * K + k0 + col;
    const unsigned short* gb = Bt + (size_t)(n0 + row) * K + k0 + col;
    unsigned int la = ldsA + (unsigned int)c * 16u;
    unsigned int lb = ldsB + (unsigned int)c * 16u;
    asm volatile("global_load_async_to_lds_b128 %0, %1, off"
                 :: "v"(la), "v"(ga) : "memory");
    asm volatile("global_load_async_to_lds_b128 %0, %1, off"
                 :: "v"(lb), "v"(gb) : "memory");
  }
}

__global__ __launch_bounds__(128)
void k_gemm_bf16_lds(const unsigned short* __restrict__ A,
                     const unsigned short* __restrict__ Bt,
                     float* __restrict__ C,
                     const float* __restrict__ bias,
                     const float* __restrict__ addsrc,
                     int M, int N, int K) {
  __shared__ unsigned short lA[2][TM * TK];   // 2 x 4 KB
  __shared__ unsigned short lB[2][TN * TK];   // 2 x 4 KB

  const int tid  = threadIdx.x;
  const int lane = tid & 31;
  const int wave = tid >> 5;
  const int hf   = lane >> 4;
  const int l16  = lane & 15;
  const int wi   = (wave >> 1) * 32;       // wave m-offset inside block tile
  const int wj   = (wave & 1) * 32;        // wave n-offset inside block tile

  const int n0 = blockIdx.x * TN;
  const int m0 = blockIdx.y * TM;

  // LDS byte addresses (generic shared pointer low 32 bits == LDS address)
  const unsigned int ldsA0 = (unsigned int)(uintptr_t)(&lA[0][0]);
  const unsigned int ldsB0 = (unsigned int)(uintptr_t)(&lB[0][0]);

  v8f zero = {0.f, 0.f, 0.f, 0.f, 0.f, 0.f, 0.f, 0.f};
  v8f acc[2][2];
#pragma unroll
  for (int i = 0; i < 2; ++i)
#pragma unroll
    for (int j = 0; j < 2; ++j) acc[i][j] = zero;

  // Prologue: stage tile 0 into buffer 0.
  stage_tiles(A, Bt, K, m0, n0, 0, tid, ldsA0, ldsB0);

  int buf = 0;
  for (int k0 = 0; k0 < K; k0 += TK) {
    // All waves finished reading buf^1 (tile k0-TK) -> safe to overwrite it.
    __syncthreads();

    if (k0 + TK < K) {
      // Kick off tile k0+TK into the other buffer, then wait only until the
      // current tile's 4 async ops (issued earlier, in-order) have landed.
      stage_tiles(A, Bt, K, m0, n0, k0 + TK, tid,
                  ldsA0 + (unsigned int)((buf ^ 1) * (TM * TK * 2)),
                  ldsB0 + (unsigned int)((buf ^ 1) * (TN * TK * 2)));
      asm volatile("s_wait_asynccnt 4" ::: "memory");
    } else {
      asm volatile("s_wait_asynccnt 0" ::: "memory");
    }
    __syncthreads();   // all waves' quarters of the current tile visible

    const unsigned short* curA = &lA[buf][0];
    const unsigned short* curB = &lB[buf][0];

    // Fragments from LDS (ds_load_b128). A 16x32 layout:
    //   k = (v<4?0:16) + hf*8 + (v&3)*2  -> two contiguous 16B runs per lane.
    // B 32x16 layout: same k order as A  -> two contiguous 16B runs per lane.
    FragU a[2], b[2];
#pragma unroll
    for (int i = 0; i < 2; ++i) {
      const unsigned short* p = curA + (wi + i * 16 + l16) * TK + hf * 8;
      a[i].q[0] = *(const u32x4*)(p);
      a[i].q[1] = *(const u32x4*)(p + 16);
    }
#pragma unroll
    for (int j = 0; j < 2; ++j) {
      const unsigned short* p = curB + (wj + j * 16 + l16) * TK + hf * 8;
      b[j].q[0] = *(const u32x4*)(p);
      b[j].q[1] = *(const u32x4*)(p + 16);
    }

#pragma unroll
    for (int i = 0; i < 2; ++i)
#pragma unroll
      for (int j = 0; j < 2; ++j)
        acc[i][j] = __builtin_amdgcn_wmma_f32_16x16x32_bf16(
            false, a[i].v, false, b[j].v, (short)0, acc[i][j], false, false);

    buf ^= 1;
  }

  // Store: lane l16 = column, VGPR r -> row hf*8 + r. No guards (exact tiles).
#pragma unroll
  for (int i = 0; i < 2; ++i) {
#pragma unroll
    for (int j = 0; j < 2; ++j) {
      int n = n0 + wj + j * 16 + l16;
      float bv = bias ? bias[n] : 0.f;
#pragma unroll
      for (int r = 0; r < 8; ++r) {
        int m = m0 + wi + i * 16 + hf * 8 + r;
        float v = acc[i][j][r] + bv;
        if (addsrc) v += addsrc[(size_t)m * N + n];
        C[(size_t)m * N + n] = v;
      }
    }
  }
}

// ---------------------------------------------------------------------------
// Guarded global-direct WMMA GEMM (used only for the 16x1000x512 head).
// One wave per 32x32 tile.
// ---------------------------------------------------------------------------
__global__ __launch_bounds__(32)
void k_gemm_bf16(const unsigned short* __restrict__ A,
                 const unsigned short* __restrict__ Bt,
                 float* __restrict__ C,
                 const float* __restrict__ bias,
                 const float* __restrict__ addsrc,
                 int M, int N, int K) {
  const int lane = threadIdx.x & 31;
  const int hf   = lane >> 4;
  const int l16  = lane & 15;
  const int n0   = blockIdx.x * 32;
  const int m0   = blockIdx.y * 32;

  v8f zero = {0.f, 0.f, 0.f, 0.f, 0.f, 0.f, 0.f, 0.f};
  v8f acc[2][2];
#pragma unroll
  for (int i = 0; i < 2; ++i)
#pragma unroll
    for (int j = 0; j < 2; ++j) acc[i][j] = zero;

  const u32x4 zq = {0u, 0u, 0u, 0u};

  for (int k0 = 0; k0 < K; k0 += 32) {
    FragU a[2], b[2];
#pragma unroll
    for (int i = 0; i < 2; ++i) {
      int m = m0 + i * 16 + l16;
      if (m < M) {
        const unsigned short* p = A + (size_t)m * K + k0 + hf * 8;
        a[i].q[0] = *(const u32x4*)(p);
        a[i].q[1] = *(const u32x4*)(p + 16);
      } else {
        a[i].q[0] = zq;
        a[i].q[1] = zq;
      }
    }
#pragma unroll
    for (int j = 0; j < 2; ++j) {
      int n = n0 + j * 16 + l16;
      if (n < N) {
        const unsigned short* p = Bt + (size_t)n * K + k0 + hf * 8;
        b[j].q[0] = *(const u32x4*)(p);
        b[j].q[1] = *(const u32x4*)(p + 16);
      } else {
        b[j].q[0] = zq;
        b[j].q[1] = zq;
      }
    }

#pragma unroll
    for (int i = 0; i < 2; ++i)
#pragma unroll
      for (int j = 0; j < 2; ++j)
        acc[i][j] = __builtin_amdgcn_wmma_f32_16x16x32_bf16(
            false, a[i].v, false, b[j].v, (short)0, acc[i][j], false, false);
  }

#pragma unroll
  for (int i = 0; i < 2; ++i) {
#pragma unroll
    for (int j = 0; j < 2; ++j) {
      int n = n0 + j * 16 + l16;
      if (n >= N) continue;
      float bv = bias ? bias[n] : 0.f;
#pragma unroll
      for (int r = 0; r < 8; ++r) {
        int m = m0 + i * 16 + hf * 8 + r;
        if (m < M) {
          float v = acc[i][j][r] + bv;
          if (addsrc) v += addsrc[(size_t)m * N + n];
          C[(size_t)m * N + n] = v;
        }
      }
    }
  }
}

// ---------------------------------------------------------------------------
// Patchify:  x[16,224,224,3] -> bf16 patches [3136, 768]
// ---------------------------------------------------------------------------
__global__ void k_patchify(const float* __restrict__ x,
                           unsigned short* __restrict__ dst) {
  size_t idx = (size_t)blockIdx.x * blockDim.x + threadIdx.x;
  const size_t total = (size_t)MTOK * PATCHD;
  if (idx >= total) return;
  int c = (int)(idx % PATCHD);
  int r = (int)(idx / PATCHD);
  int b = r / LL, l = r % LL;
  int py = l / NPS, px = l % NPS;
  int ch = c % 3;
  int pw = (c / 3) % PHH;
  int ph = (c / 3) / PHH;
  int hh = py * PHH + ph;
  int ww = px * PHH + pw;
  float v = x[(((size_t)b * 224 + hh) * 224 + ww) * 3 + ch];
  dst[idx] = f2bf(v);
}

// Transpose-pack weight [K,N] f32 -> [N,K] bf16
__global__ void k_transpack(const float* __restrict__ w,
                            unsigned short* __restrict__ wt, int K, int N) {
  size_t idx = (size_t)blockIdx.x * blockDim.x + threadIdx.x;
  size_t total = (size_t)K * N;
  if (idx >= total) return;
  int k = (int)(idx % K);
  int n = (int)(idx / K);
  wt[idx] = f2bf(w[(size_t)k * N + n]);
}

// Strided f32 -> bf16 pack (sub-matrix)
__global__ void k_pack_bf_strided(const float* __restrict__ src, int srcld,
                                  unsigned short* __restrict__ dst, int dstld,
                                  int rows, int cols) {
  size_t idx = (size_t)blockIdx.x * blockDim.x + threadIdx.x;
  size_t total = (size_t)rows * cols;
  if (idx >= total) return;
  int c = (int)(idx % cols);
  int r = (int)(idx / cols);
  dst[(size_t)r * dstld + c] = f2bf(src[(size_t)r * srcld + c]);
}

// Add 2D sincos positional embedding (computed on the fly) to h[3136,512]
__global__ void k_posemb_add(float* __restrict__ h) {
  int idx = blockIdx.x * blockDim.x + threadIdx.x;
  if (idx >= MTOK * DM) return;
  int c = idx % DM;
  int r = idx / DM;
  int l = r % LL;
  int yy = l / NPS, xx = l % NPS;
  int quad = c / (DM / 4);        // 0:sin(x) 1:cos(x) 2:sin(y) 3:cos(y)
  int j = c % (DM / 4);
  float omega = __powf(10000.0f, -(float)j / (float)(DM / 4 - 1));
  float pos = (quad < 2) ? (float)xx : (float)yy;
  float ang = pos * omega;
  h[idx] += (quad & 1) ? __cosf(ang) : __sinf(ang);
}

// RMSNorm per row (one wave32 per row), output directly as bf16 GEMM operand
__global__ __launch_bounds__(32)
void k_rmsnorm_bf(const float* __restrict__ h, const float* __restrict__ w,
                  unsigned short* __restrict__ out) {
  int row = blockIdx.x;
  int lane = threadIdx.x;
  const float* hr = h + (size_t)row * DM;
  float ss = 0.f;
  for (int c = lane; c < DM; c += 32) { float v = hr[c]; ss += v * v; }
#pragma unroll
  for (int off = 16; off > 0; off >>= 1) ss += __shfl_xor(ss, off, 32);
  float inv = rsqrtf(ss / (float)DM + 1e-5f);
  unsigned short* orow = out + (size_t)row * DM;
  for (int c = lane; c < DM; c += 32) orow[c] = f2bf(hr[c] * inv * w[c]);
}

// Causal depthwise conv (k=4) over sequence + SiLU; reads xi = xr[:, :DI]
__global__ void k_conv_silu(const float* __restrict__ xr,
                            const float* __restrict__ cw,
                            float* __restrict__ xcf,
                            unsigned short* __restrict__ xcb) {
  int idx = blockIdx.x * blockDim.x + threadIdx.x;
  if (idx >= MTOK * DI) return;
  int d = idx % DI;
  int r = idx / DI;
  int b = r / LL, l = r % LL;
  float s = 0.f;
#pragma unroll
  for (int k = 0; k < NCONV; ++k) {
    int ls = l - (NCONV - 1) + k;
    if (ls >= 0)
      s += xr[((size_t)b * LL + ls) * (2 * DI) + d] * cw[d * NCONV + k];
  }
  float sl = s / (1.f + __expf(-s));
  xcf[idx] = sl;
  xcb[idx] = f2bf(sl);
}

// delta = softplus(delta_raw + dt_b[col]) in place
__global__ void k_softplus_bias(float* __restrict__ dlt,
                                const float* __restrict__ bias) {
  int idx = blockIdx.x * blockDim.x + threadIdx.x;
  if (idx >= MTOK * DI) return;
  float v = dlt[idx] + bias[idx % DI];
  dlt[idx] = (v > 20.f) ? v : log1pf(__expf(v));
}

// Selective scan: one thread per (b, d) channel, 16 states in registers
__global__ void k_scan(const float* __restrict__ u,      // [MTOK,DI]
                       const float* __restrict__ dlt,    // [MTOK,DI]
                       const float* __restrict__ A_log,  // [DI,DSTATE]
                       const float* __restrict__ xdbl,   // [MTOK,64]: B@32, C@48
                       const float* __restrict__ Dv,     // [DI]
                       float* __restrict__ y) {          // [MTOK,DI]
  int idx = blockIdx.x * blockDim.x + threadIdx.x;
  if (idx >= BB * DI) return;
  int d = idx % DI;
  int b = idx / DI;
  float Av[DSTATE], s[DSTATE];
#pragma unroll
  for (int n = 0; n < DSTATE; ++n) {
    Av[n] = -__expf(A_log[d * DSTATE + n]);
    s[n] = 0.f;
  }
  float Dd = Dv[d];
  for (int l = 0; l < LL; ++l) {
    size_t row = (size_t)b * LL + l;
    float del = dlt[row * DI + d];
    float uu  = u[row * DI + d];
    const float* Bm = xdbl + row * 64 + DTR;
    const float* Cm = Bm + DSTATE;
    float du = del * uu;
    float acc = 0.f;
#pragma unroll
    for (int n = 0; n < DSTATE; ++n) {
      s[n] = __expf(del * Av[n]) * s[n] + du * Bm[n];
      acc += s[n] * Cm[n];
    }
    y[row * DI + d] = acc + uu * Dd;
  }
}

// y *= silu(res) where res = xr[:, DI:2*DI]; output bf16 for out_proj GEMM
__global__ void k_gate(const float* __restrict__ y,
                       const float* __restrict__ xr,
                       unsigned short* __restrict__ yb) {
  int idx = blockIdx.x * blockDim.x + threadIdx.x;
  if (idx >= MTOK * DI) return;
  int d = idx % DI;
  size_t r = (size_t)(idx / DI);
  float res = xr[r * (2 * DI) + DI + d];
  float g = res / (1.f + __expf(-res));
  yb[idx] = f2bf(y[idx] * g);
}

// pooled[b, c] = mean_l h[b*196+l, c]
__global__ void k_mean_pool(const float* __restrict__ h,
                            float* __restrict__ pooled) {
  int idx = blockIdx.x * blockDim.x + threadIdx.x;
  if (idx >= BB * DM) return;
  int c = idx % DM;
  int b = idx / DM;
  float s = 0.f;
  for (int l = 0; l < LL; ++l) s += h[((size_t)b * LL + l) * DM + c];
  pooled[idx] = s / (float)LL;
}

// ---------------------------------------------------------------------------
static inline int cdiv(int a, int b) { return (a + b - 1) / b; }

extern "C" void kernel_launch(void* const* d_in, const int* in_sizes, int n_in,
                              void* d_out, int out_size, void* d_ws, size_t ws_size,
                              hipStream_t stream) {
  (void)in_sizes; (void)n_in; (void)out_size; (void)ws_size;
  const float* x         = (const float*)d_in[0];
  const float* patch_w   = (const float*)d_in[1];
  const float* patch_b   = (const float*)d_in[2];
  const float* norm_w    = (const float*)d_in[3];
  const float* in_proj_w = (const float*)d_in[4];
  const float* conv_w    = (const float*)d_in[5];
  const float* x_proj_w  = (const float*)d_in[6];
  const float* dt_proj_w = (const float*)d_in[7];
  const float* dt_proj_b = (const float*)d_in[8];
  const float* A_log     = (const float*)d_in[9];
  const float* Dv        = (const float*)d_in[10];
  const float* out_proj_w= (const float*)d_in[11];
  const float* head_w    = (const float*)d_in[12];
  const float* head_b    = (const float*)d_in[13];
  float* out = (float*)d_out;

  // ---- workspace layout (256B aligned slices) ----
  char* base = (char*)d_ws;
  size_t off = 0;
  auto alloc = [&](size_t bytes) -> char* {
    char* p = base + off;
    off += (bytes + 255) & ~(size_t)255;
    return p;
  };
  float*          hs      = (float*)alloc((size_t)MTOK * DM * 4);        // residual stream
  float*          xr      = (float*)alloc((size_t)MTOK * 2 * DI * 4);    // in_proj out
  unsigned short* patches = (unsigned short*)xr;                         // alias (used before xr)
  unsigned short* xn_bf   = (unsigned short*)alloc((size_t)MTOK * DM * 2);
  float*          xcf     = (float*)alloc((size_t)MTOK * DI * 4);
  unsigned short* xcb     = (unsigned short*)alloc((size_t)MTOK * DI * 2);
  unsigned short* y_bf    = xcb;                                         // alias (xcb dead by then)
  float*          xdbl    = (float*)alloc((size_t)MTOK * 64 * 4);
  unsigned short* dta_bf  = (unsigned short*)alloc((size_t)MTOK * DTR * 2);
  float*          delta   = (float*)alloc((size_t)MTOK * DI * 4);
  float*          yv      = (float*)alloc((size_t)MTOK * DI * 4);
  float*          pooled  = (float*)alloc((size_t)BB * DM * 4);
  unsigned short* pool_bf = (unsigned short*)alloc((size_t)BB * DM * 2);
  unsigned short* wt_patch= (unsigned short*)alloc((size_t)DM * PATCHD * 2);
  unsigned short* wt_in   = (unsigned short*)alloc((size_t)(2 * DI) * DM * 2);
  unsigned short* wt_xp   = (unsigned short*)alloc((size_t)(DTR + 2 * DSTATE) * DI * 2);
  unsigned short* wt_dt   = (unsigned short*)alloc((size_t)DI * DTR * 2);
  unsigned short* wt_out  = (unsigned short*)alloc((size_t)DM * DI * 2);
  unsigned short* wt_head = (unsigned short*)alloc((size_t)NCLASSES * DM * 2);

  const int EW = 256;   // elementwise block size (8 wave32s)

  // ---- patch embedding ----
  {
    size_t n = (size_t)MTOK * PATCHD;
    k_patchify<<<dim3((unsigned)((n + EW - 1) / EW)), dim3(EW), 0, stream>>>(x, patches);
    size_t wn = (size_t)PATCHD * DM;
    k_transpack<<<dim3((unsigned)((wn + EW - 1) / EW)), dim3(EW), 0, stream>>>(patch_w, wt_patch, PATCHD, DM);
    k_gemm_bf16_lds<<<dim3(DM / 64, MTOK / 64), dim3(128), 0, stream>>>(
        patches, wt_patch, hs, patch_b, nullptr, MTOK, DM, PATCHD);
    int pn = MTOK * DM;
    k_posemb_add<<<dim3(cdiv(pn, EW)), dim3(EW), 0, stream>>>(hs);
  }

  // ---- Mamba layers ----
  for (int i = 0; i < NLAYERS; ++i) {
    const float* w_in  = in_proj_w  + (size_t)i * DM * 2 * DI;
    const float* w_cv  = conv_w     + (size_t)i * DI * NCONV;
    const float* w_xp  = x_proj_w   + (size_t)i * DI * (DTR + 2 * DSTATE);
    const float* w_dt  = dt_proj_w  + (size_t)i * DTR * DI;
    const float* b_dt  = dt_proj_b  + (size_t)i * DI;
    const float* Alog_i= A_log      + (size_t)i * DI * DSTATE;
    const float* Dv_i  = Dv         + (size_t)i * DI;
    const float* w_out = out_proj_w + (size_t)i * DI * DM;
    const float* w_nm  = norm_w     + (size_t)i * DM;

    // pack layer weights (transposed) to bf16
    {
      size_t n1 = (size_t)DM * 2 * DI;
      k_transpack<<<dim3((unsigned)((n1 + EW - 1) / EW)), dim3(EW), 0, stream>>>(w_in, wt_in, DM, 2 * DI);
      size_t n2 = (size_t)DI * (DTR + 2 * DSTATE);
      k_transpack<<<dim3((unsigned)((n2 + EW - 1) / EW)), dim3(EW), 0, stream>>>(w_xp, wt_xp, DI, DTR + 2 * DSTATE);
      size_t n3 = (size_t)DTR * DI;
      k_transpack<<<dim3((unsigned)((n3 + EW - 1) / EW)), dim3(EW), 0, stream>>>(w_dt, wt_dt, DTR, DI);
      size_t n4 = (size_t)DI * DM;
      k_transpack<<<dim3((unsigned)((n4 + EW - 1) / EW)), dim3(EW), 0, stream>>>(w_out, wt_out, DI, DM);
    }

    // RMSNorm -> bf16
    k_rmsnorm_bf<<<dim3(MTOK), dim3(32), 0, stream>>>(hs, w_nm, xn_bf);

    // in_proj: [3136,512] x [512,2048] -> xr
    k_gemm_bf16_lds<<<dim3(2 * DI / 64, MTOK / 64), dim3(128), 0, stream>>>(
        xn_bf, wt_in, xr, nullptr, nullptr, MTOK, 2 * DI, DM);

    // causal conv + SiLU
    k_conv_silu<<<dim3(cdiv(MTOK * DI, EW)), dim3(EW), 0, stream>>>(xr, w_cv, xcf, xcb);

    // x_proj: [3136,1024] x [1024,64] -> xdbl
    k_gemm_bf16_lds<<<dim3((DTR + 2 * DSTATE) / 64, MTOK / 64), dim3(128), 0, stream>>>(
        xcb, wt_xp, xdbl, nullptr, nullptr, MTOK, DTR + 2 * DSTATE, DI);

    // dt portion -> bf16
    k_pack_bf_strided<<<dim3(cdiv(MTOK * DTR, EW)), dim3(EW), 0, stream>>>(
        xdbl, DTR + 2 * DSTATE, dta_bf, DTR, MTOK, DTR);

    // dt_proj: [3136,32] x [32,1024] -> delta
    k_gemm_bf16_lds<<<dim3(DI / 64, MTOK / 64), dim3(128), 0, stream>>>(
        dta_bf, wt_dt, delta, nullptr, nullptr, MTOK, DI, DTR);

    // softplus(delta + dt_b)
    k_softplus_bias<<<dim3(cdiv(MTOK * DI, EW)), dim3(EW), 0, stream>>>(delta, b_dt);

    // selective scan
    k_scan<<<dim3(cdiv(BB * DI, EW)), dim3(EW), 0, stream>>>(xcf, delta, Alog_i, xdbl, Dv_i, yv);

    // gate with SiLU(res) -> bf16
    k_gate<<<dim3(cdiv(MTOK * DI, EW)), dim3(EW), 0, stream>>>(yv, xr, y_bf);

    // out_proj + residual add into hs
    k_gemm_bf16_lds<<<dim3(DM / 64, MTOK / 64), dim3(128), 0, stream>>>(
        y_bf, wt_out, hs, nullptr, hs, MTOK, DM, DI);
  }

  // ---- head ----
  k_mean_pool<<<dim3(cdiv(BB * DM, EW)), dim3(EW), 0, stream>>>(hs, pooled);
  k_pack_bf_strided<<<dim3(cdiv(BB * DM, EW)), dim3(EW), 0, stream>>>(
      pooled, DM, pool_bf, DM, BB, DM);
  {
    size_t wn = (size_t)DM * NCLASSES;
    k_transpack<<<dim3((unsigned)((wn + EW - 1) / EW)), dim3(EW), 0, stream>>>(head_w, wt_head, DM, NCLASSES);
  }
  k_gemm_bf16<<<dim3(cdiv(NCLASSES, 32), cdiv(BB, 32)), dim3(32), 0, stream>>>(
      pool_bf, wt_head, out, head_b, nullptr, BB, NCLASSES, DM);
}
